// VqLayer_16518444220548
// MI455X (gfx1250) — matrix operand order repaired and checked by reading, not compiled, not run on hardware
//
#include <hip/hip_runtime.h>

// VQ layer fused kernel for gfx1250 (MI455X).
//   x:[8,4096,256] f32, embedding:[4096,256] f32
//   out: output_vector [8*4096*256] f32  ++  vq_loss [1] f32
//
// Key identity (log-softmax weights are linear in scores):
//   weights@E = 2*x*(E^T E) - (||x||^2 + logsumexp(s))*sumE - c
// so the second big GEMM collapses into a 256x256 epilogue GEMM.
//
// This revision: A-fragments hoisted into registers for the whole kernel, and
// the E-chunk staging is double-buffered with GLOBAL_LOAD_ASYNC_TO_LDS_B128
// (ASYNCcnt) so the next chunk's DMA overlaps the current chunk's WMMAs.

typedef __attribute__((ext_vector_type(16))) __bf16 v16bf;
typedef __attribute__((ext_vector_type(8)))  __bf16 v8bf;
typedef __attribute__((ext_vector_type(8)))  float  v8f;

#define D_DIM    256
#define K_CODES  4096
#define M_TILE   128   // rows of x per workgroup
#define N_CHUNK  64    // codes per streamed chunk
#define XSTRIDE  264   // LDS row stride in bf16 (256 + 8 pad; 132 dwords = 4 mod 64 banks)

// ---- LDS layout (byte offsets; kernel has no static __shared__, dynamic LDS starts at 0)
#define XS_BYTES   (M_TILE * XSTRIDE * 2)          // 67584
#define ES_BUF_B   (N_CHUNK * XSTRIDE * 2)         // 33792
#define ES_OFF     (XS_BYTES)                      // two buffers back to back
#define ENORM_OFF  (ES_OFF + 2 * ES_BUF_B)         // 4096 f32
#define XN_OFF     (ENORM_OFF + K_CODES * 4)       // 128 f32
#define ROWL_OFF   (XN_OFF + M_TILE * 4)           // 128 f32
#define RBEST_OFF  (ROWL_OFF + M_TILE * 4)         // 128 i32
#define SUME_OFF   (RBEST_OFF + M_TILE * 4)        // 256 f32
#define CVEC_OFF   (SUME_OFF + D_DIM * 4)          // 256 f32
#define WLOSS_OFF  (CVEC_OFF + D_DIM * 4)          // 8 f32
#define LDS_TOTAL  (WLOSS_OFF + 8 * 4)             // 155168 B

__device__ __forceinline__ v16bf cat8(v8bf lo, v8bf hi) {
  return __builtin_shufflevector(lo, hi, 0,1,2,3,4,5,6,7,8,9,10,11,12,13,14,15);
}
// B fragment: 16 contiguous bf16 (elements 0..15 = K ascending)
__device__ __forceinline__ v16bf ldB(const __bf16* p) {
  return cat8(*(const v8bf*)p, *(const v8bf*)(p + 8));
}
// A fragment: elements 0..7 at p, elements 8..15 at p+16 (per ISA 16-bit A layout)
__device__ __forceinline__ v16bf ldA(const __bf16* p) {
  return cat8(*(const v8bf*)p, *(const v8bf*)(p + 16));
}

__device__ __forceinline__ void wait_async0() {
#if __has_builtin(__builtin_amdgcn_s_wait_asynccnt)
  __builtin_amdgcn_s_wait_asynccnt(0);
#else
  asm volatile("s_wait_asynccnt 0x0" ::: "memory");
#endif
}

// async copy one 16B packet per lane: LDS[lds_off] = MEM[gaddr], ASYNCcnt-tracked
__device__ __forceinline__ void async_ld_b128(unsigned lds_off, const void* gptr) {
  asm volatile("global_load_async_to_lds_b128 %0, %1, off"
               :: "v"(lds_off), "v"((unsigned long long)gptr)
               : "memory");
}

// stage one 64x256 bf16 E-chunk into an LDS buffer (stride XSTRIDE), async
__device__ __forceinline__ void es_async_copy(const __bf16* __restrict__ ebf, int chunk,
                                              unsigned buf_off, int t) {
  const int er = t >> 2, q = t & 3;  // 4 threads per row, 128 B each
  unsigned lds = buf_off + (unsigned)(er * XSTRIDE + q * 64) * 2;
  const __bf16* g = ebf + (size_t)(chunk * N_CHUNK + er) * D_DIM + q * 64;
  #pragma unroll
  for (int j = 0; j < 8; ++j) {
    async_ld_b128(lds, (const void*)g);
    lds += 16;
    g += 8;
  }
}

// ---------- prep 1: E -> bf16, enorm2[k] = sum_d E[k,d]^2 ----------
__global__ __launch_bounds__(256) void k_prep_e(const float* __restrict__ E,
                                                unsigned short* __restrict__ ebf_u,
                                                float* __restrict__ enorm2) {
  __bf16* ebf = reinterpret_cast<__bf16*>(ebf_u);
  const int row = blockIdx.x, t = threadIdx.x;
  float v = E[(size_t)row * D_DIM + t];
  ebf[(size_t)row * D_DIM + t] = (__bf16)v;
  float s = v * v;
  #pragma unroll
  for (int off = 1; off < 32; off <<= 1) s += __shfl_xor(s, off, 32);
  __shared__ float wsum[8];
  if ((t & 31) == 0) wsum[t >> 5] = s;
  __syncthreads();
  if (t == 0) {
    float tot = 0.f;
    #pragma unroll
    for (int i = 0; i < 8; ++i) tot += wsum[i];
    enorm2[row] = tot;
  }
}

// ---------- prep 2: sumE[d] = sum_k E[k,d];  c[d] = sum_k enorm2[k]*E[k,d] ----------
__global__ __launch_bounds__(256) void k_prep_cols(const float* __restrict__ E,
                                                   const float* __restrict__ enorm2,
                                                   float* __restrict__ sumE,
                                                   float* __restrict__ cvec) {
  const int d = blockIdx.x, t = threadIdx.x;
  float s1 = 0.f, s2 = 0.f;
  for (int k = t; k < K_CODES; k += 256) {
    float e = E[(size_t)k * D_DIM + d];
    s1 += e;
    s2 += enorm2[k] * e;
  }
  #pragma unroll
  for (int off = 1; off < 32; off <<= 1) {
    s1 += __shfl_xor(s1, off, 32);
    s2 += __shfl_xor(s2, off, 32);
  }
  __shared__ float r1[8], r2[8];
  if ((t & 31) == 0) { r1[t >> 5] = s1; r2[t >> 5] = s2; }
  __syncthreads();
  if (t == 0) {
    float a = 0.f, b = 0.f;
    #pragma unroll
    for (int i = 0; i < 8; ++i) { a += r1[i]; b += r2[i]; }
    sumE[d] = a;
    cvec[d] = b;
  }
}

// ---------- prep 3: Gbf = bf16(E^T E), fp32 accumulate ----------
__global__ __launch_bounds__(256) void k_gram(const float* __restrict__ E,
                                              unsigned short* __restrict__ Gbf_u) {
  __bf16* Gb = reinterpret_cast<__bf16*>(Gbf_u);
  __shared__ float As[32][33];
  __shared__ float Bs[32][33];
  const int i0 = blockIdx.x * 32, j0 = blockIdx.y * 32;
  const int tx = threadIdx.x & 15, ty = threadIdx.x >> 4;
  float acc[2][2] = {{0.f, 0.f}, {0.f, 0.f}};
  for (int kb = 0; kb < K_CODES; kb += 32) {
    #pragma unroll
    for (int s = 0; s < 4; ++s) {
      int idx = threadIdx.x + s * 256;
      int kk = idx >> 5, ii = idx & 31;
      As[kk][ii] = E[(size_t)(kb + kk) * D_DIM + i0 + ii];
      Bs[kk][ii] = E[(size_t)(kb + kk) * D_DIM + j0 + ii];
    }
    __syncthreads();
    #pragma unroll 8
    for (int kk = 0; kk < 32; ++kk) {
      float a0 = As[kk][2 * ty], a1 = As[kk][2 * ty + 1];
      float b0 = Bs[kk][2 * tx], b1 = Bs[kk][2 * tx + 1];
      acc[0][0] += a0 * b0; acc[0][1] += a0 * b1;
      acc[1][0] += a1 * b0; acc[1][1] += a1 * b1;
    }
    __syncthreads();
  }
  #pragma unroll
  for (int a = 0; a < 2; ++a)
    #pragma unroll
    for (int b = 0; b < 2; ++b)
      Gb[(size_t)(i0 + 2 * ty + a) * D_DIM + (j0 + 2 * tx + b)] = (__bf16)acc[a][b];
}

// ---------- main fused kernel ----------
__global__ __launch_bounds__(256) void k_vq_main(
    const float* __restrict__ X, const float* __restrict__ E,
    const unsigned short* __restrict__ ebf_u, const unsigned short* __restrict__ Gbf_u,
    const float* __restrict__ enorm2, const float* __restrict__ sumE,
    const float* __restrict__ cvec, float* __restrict__ Out,
    float* __restrict__ partial) {
  const __bf16* ebf = reinterpret_cast<const __bf16*>(ebf_u);
  const __bf16* Gbf = reinterpret_cast<const __bf16*>(Gbf_u);

  extern __shared__ char smem[];
  __bf16* Xs        = (__bf16*)(smem);
  float*  enorm_all = (float*)(smem + ENORM_OFF);
  float*  xnorm2_s  = (float*)(smem + XN_OFF);
  float*  rowL_s    = (float*)(smem + ROWL_OFF);
  int*    rowBest_s = (int*)(smem + RBEST_OFF);
  float*  sumE_s    = (float*)(smem + SUME_OFF);
  float*  cvec_s    = (float*)(smem + CVEC_OFF);
  float*  wloss     = (float*)(smem + WLOSS_OFF);

  const int t    = threadIdx.x;
  const int w    = t >> 5;
  const int lane = t & 31;
  const int half = lane >> 4;     // K-half select for A/B fragments
  const int l15  = lane & 15;     // N (column) within 16-wide tile
  const int rowBase = blockIdx.x * M_TILE;
  const int rA = 16 * w;          // this wave's 16-row block within the tile

  sumE_s[t] = sumE[t];
  cvec_s[t] = cvec[t];
  #pragma unroll
  for (int i = 0; i < K_CODES / 256; ++i)
    enorm_all[t + 256 * i] = enorm2[t + 256 * i];

  // ---- load X tile: fp32 -> bf16 in LDS, compute xnorm2 per row ----
  {
    const int r = t >> 1, seg = t & 1;
    const float* xp = X + (size_t)(rowBase + r) * D_DIM + seg * 128;
    __bf16* dst = Xs + r * XSTRIDE + seg * 128;
    float ss = 0.f;
    #pragma unroll
    for (int j = 0; j < 16; ++j) {
      float4 a = ((const float4*)xp)[2 * j];
      float4 b = ((const float4*)xp)[2 * j + 1];
      v8bf o;
      o[0] = (__bf16)a.x; o[1] = (__bf16)a.y; o[2] = (__bf16)a.z; o[3] = (__bf16)a.w;
      o[4] = (__bf16)b.x; o[5] = (__bf16)b.y; o[6] = (__bf16)b.z; o[7] = (__bf16)b.w;
      *(v8bf*)(dst + 8 * j) = o;
      ss += a.x * a.x + a.y * a.y + a.z * a.z + a.w * a.w;
      ss += b.x * b.x + b.y * b.y + b.z * b.z + b.w * b.w;
    }
    if (seg == 0) xnorm2_s[r] = ss;
    __syncthreads();                 // Xs complete; xnorm2 seg0 written
    if (seg == 1) xnorm2_s[r] += ss; // finalized before next barrier
  }

  // ---- hoist the 8 A-fragments (whole-kernel invariant) into registers ----
  const __bf16* aRow = Xs + (rA + l15) * XSTRIDE + 16 * half;
  v16bf afrag[8];
  #pragma unroll
  for (int ks = 0; ks < 8; ++ks) afrag[ks] = ldB(aRow + 32 * ks);

  // kick off the first E-chunk DMA (async, lands behind the loop-top barrier)
  es_async_copy(ebf, 0, ES_OFF, t);

  // per-lane online stats for 8 rows (rA + i + 8*half), codes == l15 (mod 16)
  float m[8], se[8];
  int bc[8];
  #pragma unroll
  for (int i = 0; i < 8; ++i) { m[i] = -__builtin_inff(); se[i] = 0.f; bc[i] = 0; }

  // ---- stream codes: scores s = 2*x.e - ||x||^2 - ||e||^2 via WMMA ----
  for (int c = 0; c < K_CODES / N_CHUNK; ++c) {
    wait_async0();       // our chunk-c packets have landed
    __syncthreads();     // everyone's landed; buffer (c+1)&1 free for rewrite
    if (c + 1 < K_CODES / N_CHUNK)
      es_async_copy(ebf, c + 1, ES_OFF + ((unsigned)((c + 1) & 1)) * ES_BUF_B, t);

    const __bf16* EsC = (const __bf16*)(smem + ES_OFF + (unsigned)(c & 1) * ES_BUF_B);
    v8f z = {};
    v8f acc[4] = {z, z, z, z};
    const __bf16* bRow0 = EsC + ( 0 + l15) * XSTRIDE + 16 * half;
    const __bf16* bRow1 = EsC + (16 + l15) * XSTRIDE + 16 * half;
    const __bf16* bRow2 = EsC + (32 + l15) * XSTRIDE + 16 * half;
    const __bf16* bRow3 = EsC + (48 + l15) * XSTRIDE + 16 * half;
    #pragma unroll
    for (int ks = 0; ks < 8; ++ks) {
      const int k0 = 32 * ks;
      acc[0] = __builtin_amdgcn_wmma_f32_16x16x32_bf16(false, afrag[ks], false, ldB(bRow0 + k0), (short)0, acc[0], false, false);
      acc[1] = __builtin_amdgcn_wmma_f32_16x16x32_bf16(false, afrag[ks], false, ldB(bRow1 + k0), (short)0, acc[1], false, false);
      acc[2] = __builtin_amdgcn_wmma_f32_16x16x32_bf16(false, afrag[ks], false, ldB(bRow2 + k0), (short)0, acc[2], false, false);
      acc[3] = __builtin_amdgcn_wmma_f32_16x16x32_bf16(false, afrag[ks], false, ldB(bRow3 + k0), (short)0, acc[3], false, false);
    }

    // online argmax + logsumexp update (co-executes with XDL pipe)
    #pragma unroll
    for (int i = 0; i < 8; ++i) {
      const float xn = xnorm2_s[rA + i + 8 * half];
      #pragma unroll
      for (int nb = 0; nb < 4; ++nb) {
        const float s = 2.f * acc[nb][i] - xn - enorm_all[c * N_CHUNK + 16 * nb + l15];
        const int code = c * N_CHUNK + 16 * nb + l15;
        if (s > m[i]) {
          se[i] = se[i] * __expf(m[i] - s) + 1.f;
          m[i] = s;
          bc[i] = code;
        } else {
          se[i] += __expf(s - m[i]);
        }
      }
    }
  }

  // ---- merge stats across the 16 lanes sharing each row ----
  #pragma unroll
  for (int i = 0; i < 8; ++i) {
    #pragma unroll
    for (int off = 1; off < 16; off <<= 1) {
      float om  = __shfl_xor(m[i],  off, 32);
      float ose = __shfl_xor(se[i], off, 32);
      int   oc  = __shfl_xor(bc[i], off, 32);
      float nm = fmaxf(m[i], om);
      se[i] = se[i] * __expf(m[i] - nm) + ose * __expf(om - nm);
      if (om > m[i]) bc[i] = oc;
      m[i] = nm;
    }
  }
  if (l15 == 0) {
    #pragma unroll
    for (int i = 0; i < 8; ++i) {
      const int r = rA + i + 8 * half;
      rowL_s[r] = m[i] + __logf(se[i]);
      rowBest_s[r] = bc[i];
    }
  }
  __syncthreads();

  // ---- epilogue: Y = Xbf * G (WMMA, A-frags reused), fuse output + loss ----
  float lossAcc = 0.f;
  for (int jb = 0; jb < 16; ++jb) {
    const int j = 16 * jb + l15;
    v8f acc = {};
    const __bf16* gRow = Gbf + (size_t)j * D_DIM + 16 * half;  // G symmetric: row j == col j
    #pragma unroll
    for (int ks = 0; ks < 8; ++ks) {
      acc = __builtin_amdgcn_wmma_f32_16x16x32_bf16(false, afrag[ks], false, ldB(gRow + 32 * ks), (short)0, acc, false, false);
    }
    const float sE = sumE_s[j], cj = cvec_s[j];
    #pragma unroll
    for (int i = 0; i < 8; ++i) {
      const int r = rA + i + 8 * half;
      const int gr = rowBase + r;
      const float q  = E[(size_t)rowBest_s[r] * D_DIM + j];
      const float wo = 2.f * acc[i] - (xnorm2_s[r] + rowL_s[r]) * sE - cj;
      const float xv = X[(size_t)gr * D_DIM + j];
      const float df = q - xv;
      lossAcc += df * df;
      Out[(size_t)gr * D_DIM + j] = 0.5f * (wo + q);
    }
  }

  // deterministic loss reduction: wave -> block -> d_ws partial
  #pragma unroll
  for (int off = 1; off < 32; off <<= 1) lossAcc += __shfl_xor(lossAcc, off, 32);
  if (lane == 0) wloss[w] = lossAcc;
  __syncthreads();
  if (t == 0) {
    float s = 0.f;
    #pragma unroll
    for (int i = 0; i < 8; ++i) s += wloss[i];
    partial[blockIdx.x] = s;
  }
}

// ---------- final deterministic loss reduce ----------
__global__ __launch_bounds__(256) void k_loss(const float* __restrict__ partial, int n,
                                              float* __restrict__ lossOut) {
  __shared__ float sm[256];
  const int t = threadIdx.x;
  float s = 0.f;
  for (int i = t; i < n; i += 256) s += partial[i];
  sm[t] = s;
  __syncthreads();
  for (int stride = 128; stride > 0; stride >>= 1) {
    if (t < stride) sm[t] += sm[t + stride];
    __syncthreads();
  }
  // vq_loss = q_latent + 0.25*e_latent = 1.25 * mean((quantized - x)^2)
  if (t == 0) lossOut[0] = 1.25f * sm[0] / 8388608.f;
}

extern "C" void kernel_launch(void* const* d_in, const int* in_sizes, int n_in,
                              void* d_out, int out_size, void* d_ws, size_t ws_size,
                              hipStream_t stream) {
  (void)n_in; (void)ws_size;
  const float* X = (const float*)d_in[0];   // [8,4096,256] f32
  const float* E = (const float*)d_in[1];   // [4096,256]  f32
  float* Out = (float*)d_out;

  // workspace layout (needs ~2.15 MB)
  char* ws = (char*)d_ws;
  unsigned short* ebf = (unsigned short*)(ws);             // 2,097,152 B
  unsigned short* Gbf = (unsigned short*)(ws + 2097152);   //   131,072 B
  float* enorm2       = (float*)(ws + 2228224);            //    16,384 B
  float* sumE         = (float*)(ws + 2244608);            //     1,024 B
  float* cvec         = (float*)(ws + 2245632);            //     1,024 B
  float* partial      = (float*)(ws + 2246656);            //     1,024 B

  const int nRows = in_sizes[0] / D_DIM;      // 32768
  const int nBlocks = nRows / M_TILE;         // 256

  k_prep_e   <<<K_CODES, 256, 0, stream>>>(E, ebf, enorm2);
  k_prep_cols<<<D_DIM,   256, 0, stream>>>(E, enorm2, sumE, cvec);
  k_gram     <<<dim3(8, 8), 256, 0, stream>>>(E, Gbf);

  k_vq_main<<<nBlocks, 256, LDS_TOTAL, stream>>>(X, E, ebf, Gbf, enorm2, sumE, cvec, Out, partial);
  k_loss<<<1, 256, 0, stream>>>(partial, nBlocks, (float*)d_out + (out_size - 1));
}
